// MultiHeadAttention_62319975465542
// MI455X (gfx1250) — matrix operand order repaired and checked by reading, not compiled, not run on hardware
//
#include <hip/hip_runtime.h>

typedef _Float16 f16;
typedef __attribute__((ext_vector_type(16))) _Float16 v16h;
typedef __attribute__((ext_vector_type(8)))  _Float16 v8h;
typedef __attribute__((ext_vector_type(4)))  _Float16 v4h;
typedef __attribute__((ext_vector_type(8)))  float    v8f;
typedef __attribute__((ext_vector_type(4)))  unsigned u32x4;
typedef __attribute__((ext_vector_type(8)))  unsigned u32x8;

#define L_SEQ   2048
#define D_MODEL 2048
#define N_HEADS 32
#define D_HEAD  64

union FragU { v16h v; v8h h8[2]; };

static __device__ __forceinline__ v8f wmma_f16(const v16h& a, const v16h& b, const v8f& c) {
    return __builtin_amdgcn_wmma_f32_16x16x32_f16(false, a, false, b, (short)0, c, false, false);
}

// ---- DPP butterfly reductions within each 16-lane half (C-fragment rows) ----
template <int CTRL>
static __device__ __forceinline__ float dpp_mov(float v) {
    int i = __builtin_bit_cast(int, v);
    i = __builtin_amdgcn_update_dpp(0, i, CTRL, 0xF, 0xF, true);
    return __builtin_bit_cast(float, i);
}
static __device__ __forceinline__ float row16_max(float v) {
    v = fmaxf(v, dpp_mov<0xB1>(v));
    v = fmaxf(v, dpp_mov<0x4E>(v));
    v = fmaxf(v, dpp_mov<0x141>(v));
    v = fmaxf(v, dpp_mov<0x140>(v));
    return v;
}
static __device__ __forceinline__ float row16_sum(float v) {
    v += dpp_mov<0xB1>(v);
    v += dpp_mov<0x4E>(v);
    v += dpp_mov<0x141>(v);
    v += dpp_mov<0x140>(v);
    return v;
}

// ---- Tensor Data Mover: 2D tile load, f16 elements, LDS row padding 16B ----
// D# per ISA ch.8: group0 = {count, lds_addr, global_addr, type=2},
// group1 = {data_size=2B, pad_enable, pad_interval, pad_amount=4 DWORDs,
//           tensor_dim0/1, tile_dim0/1, tensor_dim0_stride}.
static __device__ __forceinline__ void tdm_load_2d(unsigned lds_addr, const void* gaddr,
                                                   unsigned td0, unsigned td1, unsigned stride0,
                                                   unsigned tile_d0, unsigned tile_d1,
                                                   unsigned pad_int) {
    uintptr_t ga = (uintptr_t)gaddr;
    u32x4 g0;
    g0[0] = 1u;                                                   // count=1
    g0[1] = lds_addr;                                             // LDS byte address
    g0[2] = (unsigned)ga;                                         // global_addr[31:0]
    g0[3] = (unsigned)((ga >> 32) & 0x01FFFFFFu) | (2u << 30);    // addr[56:32] | type=2
    u32x8 g1;
    g1[0] = (1u << 16) | (1u << 20) | (pad_int << 22) | (3u << 25);
    g1[1] = (td0 & 0xFFFFu) << 16;
    g1[2] = (td0 >> 16) | ((td1 & 0xFFFFu) << 16);
    g1[3] = (td1 >> 16) | (tile_d0 << 16);
    g1[4] = tile_d1;                                              // tile_dim2 = 0
    g1[5] = stride0;                                              // dim0 stride (elements)
    g1[6] = 0u;
    g1[7] = 0u;
    asm volatile("tensor_load_to_lds %0, %1" :: "s"(g0), "s"(g1) : "memory");
}
static __device__ __forceinline__ unsigned lds_off(const void* p) {
    return (unsigned)(uintptr_t)p;   // low 32 bits of generic LDS ptr = LDS address
}

// ---------------------------------------------------------------------------
// Prepass: fp32 -> f16 copy of X (consumed as GEMM A).
// ---------------------------------------------------------------------------
__global__ __launch_bounds__(256) void cvt_f32_f16(const float* __restrict__ src,
                                                   _Float16* __restrict__ dst) {
    size_t i = ((size_t)blockIdx.x * 256 + threadIdx.x) * 4;
    float4 f = *(const float4*)(src + i);
    v4h h = { (_Float16)f.x, (_Float16)f.y, (_Float16)f.z, (_Float16)f.w };
    *(v4h*)(dst + i) = h;
}

// Prepass: W (N x K fp32, row-major) -> Wt (K x N f16) tiled transpose.
__global__ __launch_bounds__(256) void transpose_f32_f16(const float* __restrict__ W,
                                                         _Float16* __restrict__ Wt) {
    __shared__ __align__(16) _Float16 tile[32][32 + 8];
    const int n0 = blockIdx.x * 32, k0 = blockIdx.y * 32;
    const int c  = threadIdx.x & 31;
    const int r0 = (threadIdx.x >> 5) * 4;
#pragma unroll
    for (int i = 0; i < 4; ++i)
        tile[r0 + i][c] = (_Float16)W[(size_t)(n0 + r0 + i) * D_MODEL + k0 + c];
    __syncthreads();
#pragma unroll
    for (int i = 0; i < 4; ++i)
        Wt[(size_t)(k0 + r0 + i) * D_MODEL + n0 + c] = tile[c][r0 + i];
}

// ---------------------------------------------------------------------------
// GEMM (all f16 in, f32 accum): C[m][n] = sum_k A[m][k]*Bt[k][n] + bias[n]
// TDM double-buffered staging. 256 thr (8 waves, 4x2), tile 128x128, BK=64.
// MODE 0: f16 -> Qh[h][m][d] | 1: f16 -> Kt[h][d][m] | 2: f16 -> Vh[h][m][d] | 3: f32 out
// ---------------------------------------------------------------------------
template <int MODE>
__global__ __launch_bounds__(256) void gemm16(const _Float16* __restrict__ A,
                                              const _Float16* __restrict__ Bt,
                                              const float* __restrict__ bias,
                                              void* __restrict__ out) {
    constexpr int BM = 128, BN = 128, BK = 64, PAD = 8, NIT = D_MODEL / BK;
    __shared__ __align__(16) _Float16 As[2][BM][BK + PAD];   // 2 x 18KB
    __shared__ __align__(16) _Float16 Bs[2][BK][BN + PAD];   // 2 x 17KB

    const int tid  = threadIdx.x;
    const int wave = tid >> 5;
    const int lane = tid & 31;
    const int hl   = lane >> 4;
    const int l16  = lane & 15;
    const int wm   = wave >> 1;
    const int wn   = wave & 1;
    const int Mblk = blockIdx.y * BM;
    const int Nblk = blockIdx.x * BN;
    const int kb   = l16 + 16 * hl;

    const _Float16* abase = A + (size_t)Mblk * D_MODEL;   // k advances along rows
    const _Float16* bbase = Bt + Nblk;                    // k advances down rows

    if (wave == 0) {
        tdm_load_2d(lds_off(&As[0][0][0]), abase, D_MODEL, L_SEQ, D_MODEL, BK, BM, 4);
        tdm_load_2d(lds_off(&Bs[0][0][0]), bbase, D_MODEL, D_MODEL, D_MODEL, BN, BK, 5);
    }

    v8f acc[2][4] = {};

    for (int kt = 0; kt < NIT; ++kt) {
        const int buf = kt & 1;
        if (wave == 0) {
            if (kt + 1 < NIT) {
                tdm_load_2d(lds_off(&As[buf ^ 1][0][0]), abase + (kt + 1) * BK,
                            D_MODEL, L_SEQ, D_MODEL, BK, BM, 4);
                tdm_load_2d(lds_off(&Bs[buf ^ 1][0][0]), bbase + (size_t)(kt + 1) * BK * D_MODEL,
                            D_MODEL, D_MODEL, D_MODEL, BN, BK, 5);
                __builtin_amdgcn_s_wait_tensorcnt((short)2);
            } else {
                __builtin_amdgcn_s_wait_tensorcnt((short)0);
            }
        }
        __syncthreads();

#pragma unroll
        for (int kf = 0; kf < 2; ++kf) {
            FragU a[2], b;
#pragma unroll
            for (int mf = 0; mf < 2; ++mf) {
                int m = wm * 32 + mf * 16 + l16;
                a[mf].h8[0] = *(const v8h*)&As[buf][m][kf * 32 + hl * 8];
                a[mf].h8[1] = *(const v8h*)&As[buf][m][kf * 32 + 16 + hl * 8];
            }
#pragma unroll
            for (int nf = 0; nf < 4; ++nf) {
                int n0 = wn * 64 + nf * 16;
                for (int i = 0; i < 8; ++i) {
                    b.v[i]     = Bs[buf][kf * 32 + hl * 8 + i][n0 + l16];
                    b.v[8 + i] = Bs[buf][kf * 32 + 16 + hl * 8 + i][n0 + l16];
                }
                acc[0][nf] = wmma_f16(a[0].v, b.v, acc[0][nf]);
                acc[1][nf] = wmma_f16(a[1].v, b.v, acc[1][nf]);
            }
        }
        __syncthreads();
    }

#pragma unroll
    for (int nf = 0; nf < 4; ++nf) {
        int n = Nblk + wn * 64 + nf * 16 + l16;
        float bn = bias[n];
        int hh = n & (N_HEADS - 1);
        int d  = n >> 5;
#pragma unroll
        for (int mf = 0; mf < 2; ++mf) {
            int mbase = Mblk + wm * 32 + mf * 16 + hl * 8;
#pragma unroll
            for (int r = 0; r < 8; ++r) {
                int m = mbase + r;
                float v = acc[mf][nf][r] + bn;
                if (MODE == 3) {
                    ((float*)out)[(size_t)m * D_MODEL + n] = v;
                } else {
                    _Float16 hv = (_Float16)v;
                    if (MODE == 1)
                        ((_Float16*)out)[((size_t)hh * D_HEAD + d) * L_SEQ + m] = hv;   // Kt[h][d][m]
                    else
                        ((_Float16*)out)[((size_t)hh * L_SEQ + m) * D_HEAD + d] = hv;   // Qh/Vh[h][m][d]
                }
            }
        }
    }
}

// ---------------------------------------------------------------------------
// Flash attention. Grid (32 heads, 16 q-tiles of 128 rows), 8 waves x 16 rows.
// K/V tiles TDM-staged, double-buffered. Output f16 Ofh[l][d*32+h].
// ---------------------------------------------------------------------------
__global__ __launch_bounds__(256) void attn_fa(const _Float16* __restrict__ Qh,
                                               const _Float16* __restrict__ Kt,
                                               const _Float16* __restrict__ Vh,
                                               _Float16* __restrict__ Ofh) {
    constexpr int PAD = 8, TK = 64, NIT = L_SEQ / TK;
    __shared__ __align__(16) _Float16 Ks[2][D_HEAD][TK + PAD];   // [d][key]
    __shared__ __align__(16) _Float16 Vs[2][TK][D_HEAD + PAD];   // [key][d]
    __shared__ __align__(16) _Float16 Ps[8][16][TK + PAD];       // per-wave P strip

    const int h    = blockIdx.x;
    const int qt   = blockIdx.y;
    const int tid  = threadIdx.x;
    const int wave = tid >> 5;
    const int lane = tid & 31;
    const int hl   = lane >> 4;
    const int l16  = lane & 15;
    const int kb   = l16 + 16 * hl;

    const _Float16* kbase = Kt + (size_t)h * D_HEAD * L_SEQ;   // [d][l]
    const _Float16* vbase = Vh + (size_t)h * L_SEQ * D_HEAD;   // [l][d]

    if (wave == 0) {
        tdm_load_2d(lds_off(&Ks[0][0][0]), kbase, L_SEQ, D_HEAD, L_SEQ, TK, D_HEAD, 4);
        tdm_load_2d(lds_off(&Vs[0][0][0]), vbase, D_HEAD, L_SEQ, D_HEAD, D_HEAD, TK, 4);
    }

    // Q strip (16 x 64) as two A-fragments, pre-scaled by 1/sqrt(D_MODEL)
    FragU qa[2], pa[2], bf;
    const int qrow = qt * 128 + wave * 16 + l16;
    const _Float16* qptr = Qh + ((size_t)h * L_SEQ + qrow) * D_HEAD;
    const _Float16 hscale = (_Float16)0.022097086912079608f;
#pragma unroll
    for (int kf = 0; kf < 2; ++kf) {
        qa[kf].h8[0] = *(const v8h*)(qptr + kf * 32 + hl * 8);
        qa[kf].h8[1] = *(const v8h*)(qptr + kf * 32 + 16 + hl * 8);
#pragma unroll
        for (int i = 0; i < 16; ++i) qa[kf].v[i] = qa[kf].v[i] * hscale;
    }

    float mrun[8], lrun[8];
    v8f o[4] = {};
#pragma unroll
    for (int r = 0; r < 8; ++r) { mrun[r] = -1e30f; lrun[r] = 0.0f; }

    for (int jt = 0; jt < NIT; ++jt) {
        const int buf = jt & 1;
        if (wave == 0) {
            if (jt + 1 < NIT) {
                tdm_load_2d(lds_off(&Ks[buf ^ 1][0][0]), kbase + (jt + 1) * TK,
                            L_SEQ, D_HEAD, L_SEQ, TK, D_HEAD, 4);
                tdm_load_2d(lds_off(&Vs[buf ^ 1][0][0]), vbase + (size_t)(jt + 1) * TK * D_HEAD,
                            D_HEAD, L_SEQ, D_HEAD, D_HEAD, TK, 4);
                __builtin_amdgcn_s_wait_tensorcnt((short)2);
            } else {
                __builtin_amdgcn_s_wait_tensorcnt((short)0);
            }
        }
        __syncthreads();

        // S strip (16 x 64) = Q K^T
        v8f s[4];
#pragma unroll
        for (int nf = 0; nf < 4; ++nf) {
            v8f sc = {};
#pragma unroll
            for (int kf = 0; kf < 2; ++kf) {
                for (int i = 0; i < 8; ++i) {
                    bf.v[i]     = Ks[buf][kf * 32 + hl * 8 + i][nf * 16 + l16];
                    bf.v[8 + i] = Ks[buf][kf * 32 + 16 + hl * 8 + i][nf * 16 + l16];
                }
                sc = wmma_f16(qa[kf].v, bf.v, sc);
            }
            s[nf] = sc;
        }

        // online softmax (DPP row reductions)
#pragma unroll
        for (int r = 0; r < 8; ++r) {
            float mx = fmaxf(fmaxf(s[0][r], s[1][r]), fmaxf(s[2][r], s[3][r]));
            mx = row16_max(mx);
            float mnew = fmaxf(mrun[r], mx);
            float sf = __builtin_amdgcn_exp2f((mrun[r] - mnew) * 1.44269504f);
            float rs = 0.0f;
#pragma unroll
            for (int nf = 0; nf < 4; ++nf) {
                float p = __builtin_amdgcn_exp2f((s[nf][r] - mnew) * 1.44269504f);
                s[nf][r] = p;
                rs += p;
            }
            rs = row16_sum(rs);
            lrun[r] = lrun[r] * sf + rs;
            mrun[r] = mnew;
#pragma unroll
            for (int nf = 0; nf < 4; ++nf) o[nf][r] *= sf;
        }

        // C-layout -> A-layout via wave-private LDS strip
#pragma unroll
        for (int nf = 0; nf < 4; ++nf)
#pragma unroll
            for (int r = 0; r < 8; ++r)
                Ps[wave][hl * 8 + r][nf * 16 + l16] = (_Float16)s[nf][r];
        asm volatile("s_wait_dscnt 0" ::: "memory");
#pragma unroll
        for (int kf = 0; kf < 2; ++kf) {
            pa[kf].h8[0] = *(const v8h*)&Ps[wave][l16][kf * 32 + hl * 8];
            pa[kf].h8[1] = *(const v8h*)&Ps[wave][l16][kf * 32 + 16 + hl * 8];
        }

        // O += P V
#pragma unroll
        for (int df = 0; df < 4; ++df) {
#pragma unroll
            for (int kf = 0; kf < 2; ++kf) {
                for (int i = 0; i < 8; ++i) {
                    bf.v[i]     = Vs[buf][kf * 32 + hl * 8 + i][df * 16 + l16];
                    bf.v[8 + i] = Vs[buf][kf * 32 + 16 + hl * 8 + i][df * 16 + l16];
                }
                o[df] = wmma_f16(pa[kf].v, bf.v, o[df]);
            }
        }
        __syncthreads();
    }

    // epilogue: O /= l; write f16 Ofh[m][d*32 + h]
    const int mbase = qt * 128 + wave * 16 + hl * 8;
#pragma unroll
    for (int df = 0; df < 4; ++df) {
        int d = df * 16 + l16;
        int col = d * N_HEADS + h;
#pragma unroll
        for (int r = 0; r < 8; ++r)
            Ofh[(size_t)(mbase + r) * D_MODEL + col] = (_Float16)(o[df][r] / lrun[r]);
    }
}

// ---------------------------------------------------------------------------
extern "C" void kernel_launch(void* const* d_in, const int* in_sizes, int n_in,
                              void* d_out, int out_size, void* d_ws, size_t ws_size,
                              hipStream_t stream) {
    const float* X    = (const float*)d_in[0];
    const float* Wq_w = (const float*)d_in[1];
    const float* Wq_b = (const float*)d_in[2];
    const float* Wk_w = (const float*)d_in[3];
    const float* Wk_b = (const float*)d_in[4];
    const float* Wv_w = (const float*)d_in[5];
    const float* Wv_b = (const float*)d_in[6];
    const float* Wo_w = (const float*)d_in[7];
    const float* Wo_b = (const float*)d_in[8];

    char* ws = (char*)d_ws;
    _Float16* Xh  = (_Float16*)(ws);                       // 8 MB  [l][k] f16
    _Float16* Wt  = (_Float16*)(ws + (size_t) 8 * 1048576); // 8 MB  [k][n] f16 (reused)
    _Float16* Qh  = (_Float16*)(ws + (size_t)16 * 1048576); // 8 MB  [h][l][d]
    _Float16* Kt  = (_Float16*)(ws + (size_t)24 * 1048576); // 8 MB  [h][d][l]
    _Float16* Vh  = (_Float16*)(ws + (size_t)32 * 1048576); // 8 MB  [h][l][d]
    _Float16* Ofh = (_Float16*)(ws + (size_t)40 * 1048576); // 8 MB  [l][k] f16

    dim3 blk(256);
    dim3 gcvt(D_MODEL * L_SEQ / (256 * 4));
    dim3 gtr(D_MODEL / 32, D_MODEL / 32);
    dim3 gg(D_MODEL / 128, L_SEQ / 128);

    cvt_f32_f16<<<gcvt, blk, 0, stream>>>(X, Xh);

    transpose_f32_f16<<<gtr, blk, 0, stream>>>(Wq_w, Wt);
    gemm16<0><<<gg, blk, 0, stream>>>(Xh, Wt, Wq_b, (void*)Qh);

    transpose_f32_f16<<<gtr, blk, 0, stream>>>(Wk_w, Wt);
    gemm16<1><<<gg, blk, 0, stream>>>(Xh, Wt, Wk_b, (void*)Kt);

    transpose_f32_f16<<<gtr, blk, 0, stream>>>(Wv_w, Wt);
    gemm16<2><<<gg, blk, 0, stream>>>(Xh, Wt, Wv_b, (void*)Vh);

    attn_fa<<<dim3(N_HEADS, L_SEQ / 128), blk, 0, stream>>>(Qh, Kt, Vh, Ofh);

    transpose_f32_f16<<<gtr, blk, 0, stream>>>(Wo_w, Wt);
    gemm16<3><<<gg, blk, 0, stream>>>(Ofh, Wt, Wo_b, d_out);
}
